// Attention_55860344652494
// MI455X (gfx1250) — matrix operand order repaired and checked by reading, not compiled, not run on hardware
//
#include <hip/hip_runtime.h>
#include <math.h>

// ---------------------------------------------------------------------------
// Problem constants (fixed by the reference)
// ---------------------------------------------------------------------------
#define BATCH      2
#define SEQ        2048
#define NTOK       (BATCH * SEQ)   // 4096
#define FEAT       2048
#define NH         8               // query heads
#define NKV        4               // kv heads
#define HD         256             // head dim
#define WINDOW     1024
#define K_MASK_F   (-2.3819763e+38f)
#define LN1E4_128  0.07195578442f  // ln(10000)/128

typedef __attribute__((ext_vector_type(16))) __bf16 v16bf;
typedef __attribute__((ext_vector_type(8)))  __bf16 v8bf;
typedef __attribute__((ext_vector_type(8)))  float  v8f;
typedef __attribute__((ext_vector_type(4)))  unsigned int v4u;

#define DEV static __device__ __forceinline__

// Scheduling pipeline directives (best effort; no-op if unsupported).
// Masks: 0x008 MFMA/WMMA, 0x020 VMEM read, 0x100 DS read, 0x200 DS write.
#if defined(__has_builtin)
#if __has_builtin(__builtin_amdgcn_sched_group_barrier)
#define SGB(mask, size, id) __builtin_amdgcn_sched_group_barrier((mask), (size), (id))
#endif
#endif
#ifndef SGB
#define SGB(mask, size, id)
#endif

DEV v8f vzero8() {
  v8f z;
#pragma unroll
  for (int i = 0; i < 8; ++i) z[i] = 0.0f;
  return z;
}

DEV v8f wmma_bf16(v16bf a, v16bf b, v8f c) {
  return __builtin_amdgcn_wmma_f32_16x16x32_bf16(false, a, false, b,
                                                 (short)0, c, false, false);
}

// A-matrix 16x32 bf16 fragment (ISA 7.12.2): lane<16 -> kbase 0, lane>=16 -> 8;
// halves 0..7 = K kbase..kbase+7, halves 8..15 = K kbase+16..kbase+23.
DEV v16bf frag_a(const __bf16* rowp, unsigned lane) {
  const unsigned kb = (lane & 16u) ? 8u : 0u;
  v8bf lo = *(const v8bf*)(rowp + kb);
  v8bf hi = *(const v8bf*)(rowp + kb + 16);
  return __builtin_shufflevector(lo, hi, 0,1,2,3,4,5,6,7,8,9,10,11,12,13,14,15);
}

// B-matrix 32x16 bf16 fragment: lane<16 -> K 0..7,16..23, lane>=16 -> K 8..15,24..31 of
// column lane&15.  colp points at &Btrans[col][32*kstep].
DEV v16bf frag_b(const __bf16* colp, unsigned lane) {
  const unsigned kb = (lane & 16u) ? 8u : 0u;
  v8bf lo = *(const v8bf*)(colp + kb);
  v8bf hi = *(const v8bf*)(colp + kb + 16);
  return __builtin_shufflevector(lo, hi, 0,1,2,3,4,5,6,7,8,9,10,11,12,13,14,15);
}

// ---------------------------------------------------------------------------
// Phase 0: cast f32 -> bf16 with transpose.  src: [mats][R][C] -> dst [mats][C][R]
// ---------------------------------------------------------------------------
__global__ __launch_bounds__(256) void k_cast_transpose(
    const float* __restrict__ src, __bf16* __restrict__ dst,
    int R, int C, long total) {
  long id = (long)blockIdx.x * blockDim.x + threadIdx.x;
  if (id >= total) return;
  long r = id % R;
  long rem = id / R;
  long c = rem % C;
  long m = rem / C;
  dst[id] = (__bf16)src[(m * (long)R + r) * (long)C + c];
}

// ---------------------------------------------------------------------------
// Phase 1: fused QKV projection.  grid = (NTOK/64, 16 slots), block = 128.
// slot 0..7 -> q head, 8..11 -> k head, 12..15 -> v head (written transposed).
// Double-buffered LDS, K chunk = 32; sched_group_barrier pipeline interleaves
// DS reads two fragments ahead of each WMMA.
// ---------------------------------------------------------------------------
__global__ __launch_bounds__(128) void k_qkv_gemm(
    const float* __restrict__ x,
    const __bf16* __restrict__ wq_t, const __bf16* __restrict__ wk_t,
    const __bf16* __restrict__ wv_t,
    __bf16* __restrict__ qbuf, __bf16* __restrict__ kbuf,
    __bf16* __restrict__ v_t) {
  __shared__ __bf16 aT[2][64 * 40];    // 64 tokens x 32 K (stride 40)
  __shared__ __bf16 bT[2][256 * 40];   // 256 n x 32 K (stride 40)

  const int tid = threadIdx.x;
  const unsigned lane = tid & 31u;
  const int wid = tid >> 5;
  const int cm = (int)(lane & 15u);
  const int half = (int)(lane >> 4);
  const int g0 = blockIdx.x * 64;     // flat token base
  const int slot = blockIdx.y;

  const __bf16* wt =
      (slot < 8)  ? (wq_t + (size_t)slot * HD * FEAT)
    : (slot < 12) ? (wk_t + (size_t)(slot - 8) * HD * FEAT)
                  : (wv_t + (size_t)(slot - 12) * HD * FEAT);

  v8f acc[16];
#pragma unroll
  for (int i = 0; i < 16; ++i) acc[i] = vzero8();
  const int r0 = wid * 16;

  const int arow = tid >> 1, aseg = tid & 1;   // A: 16 f32 per thread
  float fa[16];                                 // staged A (next tile)
  v4u ub[8];                                    // staged B (2 rows x 64B)

  // prologue: fetch tile 0, store into buffer 0
  {
    const float* src = x + (size_t)(g0 + arow) * FEAT + aseg * 16;
#pragma unroll
    for (int j = 0; j < 16; ++j) fa[j] = src[j];
#pragma unroll
    for (int rr = 0; rr < 2; ++rr) {
      const v4u* s = (const v4u*)(wt + (size_t)(tid * 2 + rr) * FEAT);
#pragma unroll
      for (int j = 0; j < 4; ++j) ub[rr * 4 + j] = s[j];
    }
    v8bf p0, p1;
#pragma unroll
    for (int j = 0; j < 8; ++j) p0[j] = (__bf16)fa[j];
#pragma unroll
    for (int j = 0; j < 8; ++j) p1[j] = (__bf16)fa[8 + j];
    *(v8bf*)&aT[0][arow * 40 + aseg * 16] = p0;
    *(v8bf*)&aT[0][arow * 40 + aseg * 16 + 8] = p1;
#pragma unroll
    for (int rr = 0; rr < 2; ++rr) {
      v4u* d = (v4u*)&bT[0][(tid * 2 + rr) * 40];
#pragma unroll
      for (int j = 0; j < 4; ++j) d[j] = ub[rr * 4 + j];
    }
  }

  const int NIT = FEAT / 32;   // 64
  for (int it = 0; it < NIT; ++it) {
    __syncthreads();
    const int cur = it & 1;
    // issue next tile's global loads early (latency hidden by WMMAs)
    if (it + 1 < NIT) {
      const int ko = (it + 1) * 32;
      const float* src = x + (size_t)(g0 + arow) * FEAT + ko + aseg * 16;
#pragma unroll
      for (int j = 0; j < 16; ++j) fa[j] = src[j];
#pragma unroll
      for (int rr = 0; rr < 2; ++rr) {
        const v4u* s = (const v4u*)(wt + (size_t)(tid * 2 + rr) * FEAT + ko);
#pragma unroll
        for (int j = 0; j < 4; ++j) ub[rr * 4 + j] = s[j];
      }
    }
    // compute: 16 WMMAs over the current buffer
    {
      v16bf af = frag_a(&aT[cur][(r0 + cm) * 40], lane);
#pragma unroll
      for (int nt = 0; nt < 16; ++nt) {
        v16bf bf = frag_b(&bT[cur][(nt * 16 + cm) * 40], lane);
        acc[nt] = wmma_bf16(af, bf, acc[nt]);
      }
    }
    // stash next tile into the other buffer
    if (it + 1 < NIT) {
      const int nxt = cur ^ 1;
      v8bf p0, p1;
#pragma unroll
      for (int j = 0; j < 8; ++j) p0[j] = (__bf16)fa[j];
#pragma unroll
      for (int j = 0; j < 8; ++j) p1[j] = (__bf16)fa[8 + j];
      *(v8bf*)&aT[nxt][arow * 40 + aseg * 16] = p0;
      *(v8bf*)&aT[nxt][arow * 40 + aseg * 16 + 8] = p1;
#pragma unroll
      for (int rr = 0; rr < 2; ++rr) {
        v4u* d = (v4u*)&bT[nxt][(tid * 2 + rr) * 40];
#pragma unroll
        for (int j = 0; j < 4; ++j) d[j] = ub[rr * 4 + j];
      }
    }
    // scheduling pipeline: vmem first, DS reads 1-2 fragments ahead of WMMA,
    // DS writes last.  (12 vmem reads, 34 ds reads, 16 wmma, 10 ds writes)
    SGB(0x020, 12, 0);
    SGB(0x100, 6, 0);
#pragma unroll
    for (int i = 0; i < 14; ++i) { SGB(0x008, 1, 0); SGB(0x100, 2, 0); }
    SGB(0x008, 2, 0);
    SGB(0x200, 10, 0);
  }

  // Epilogue.  C layout: lane holds col h = nt*16 + cm, rows m = r0 + v + 8*half.
  if (slot < 12) {
#pragma unroll
    for (int nt = 0; nt < 16; ++nt) {
      int h = nt * 16 + cm;
#pragma unroll
      for (int v = 0; v < 8; ++v) {
        int g = g0 + r0 + v + 8 * half;
        float val = acc[nt][v];
        if (slot < 8) {
          qbuf[((size_t)g * NH + slot) * HD + h] = (__bf16)val;
        } else {
          int b = g >> 11, t = g & (SEQ - 1), kvh = slot - 8;
          kbuf[(((size_t)(b * NKV + kvh)) * SEQ + t) * HD + h] = (__bf16)val;
        }
      }
    }
  } else {
    // V: write transposed v_t[b][kvh][h][t]; 8 consecutive t per lane -> 16B store.
    int kvh = slot - 12;
    int b = g0 >> 11;
    int tloc = (g0 & (SEQ - 1)) + r0 + 8 * half;
#pragma unroll
    for (int nt = 0; nt < 16; ++nt) {
      int h = nt * 16 + cm;
      v8bf pv;
#pragma unroll
      for (int v = 0; v < 8; ++v) pv[v] = (__bf16)acc[nt][v];
      *(v8bf*)&v_t[(((size_t)(b * NKV + kvh)) * HD + h) * SEQ + tloc] = pv;
    }
  }
}

// ---------------------------------------------------------------------------
// Phase 1b: RMS-norm + RoPE in place on q (with 1/sqrt(HD)) and k.
// One wave per (token, head-row); 8 waves / 256-thread block.
// ---------------------------------------------------------------------------
__global__ __launch_bounds__(256) void k_norm_rope(
    __bf16* __restrict__ qbuf, __bf16* __restrict__ kbuf,
    const float* __restrict__ q_scale, const float* __restrict__ k_scale,
    const int* __restrict__ seg) {
  const int tid = threadIdx.x;
  const unsigned lane = tid & 31u;
  const int wid = tid >> 5;
  const long row = (long)blockIdx.x * 8 + wid;   // 0 .. NTOK*12-1
  const int g = (int)(row / 12);
  const int slot = (int)(row % 12);

  __bf16* ptr;
  const float* sc;
  float post;
  if (slot < 8) {
    ptr = qbuf + ((size_t)g * NH + slot) * HD;
    sc = q_scale;
    post = 0.0625f;                 // HD^-0.5 = 1/16
  } else {
    int b = g >> 11, t = g & (SEQ - 1);
    ptr = kbuf + (((size_t)(b * NKV + (slot - 8))) * SEQ + t) * HD;
    sc = k_scale;
    post = 1.0f;
  }

  const int h0 = (int)lane * 8;
  v8bf raw = *(const v8bf*)(ptr + h0);
  float xv[8];
  float ss = 0.0f;
#pragma unroll
  for (int j = 0; j < 8; ++j) { xv[j] = (float)raw[j]; ss += xv[j] * xv[j]; }
#pragma unroll
  for (int m = 1; m <= 16; m <<= 1) ss += __shfl_xor(ss, m, 32);
  const float inv = rsqrtf(ss * (1.0f / (float)HD) + 1e-6f);
  const float pos = (float)seg[g];

  v8bf outp;
#pragma unroll
  for (int j = 0; j < 8; ++j) {
    float nrm = xv[j] * inv * (1.0f + sc[h0 + j]);
    float prt = __shfl_xor(nrm, 16, 32);          // partner element h +/- 128
    int i = (h0 + j) & 127;
    float ang = pos * __expf(-LN1E4_128 * (float)i);
    float sa, ca;
    __sincosf(ang, &sa, &ca);
    float o = (lane < 16u) ? (nrm * ca - prt * sa) : (nrm * ca + prt * sa);
    outp[j] = (__bf16)(o * post);
  }
  *(v8bf*)(ptr + h0) = outp;
}

// ---------------------------------------------------------------------------
// Phase 2: flash attention.  grid = (SEQ/64, NH, BATCH), block = 128 (4 waves,
// 16 query rows each).  S blocks of 32 within [t0-1023, t0+63].
// ---------------------------------------------------------------------------
__global__ __launch_bounds__(128) void k_attn(
    const __bf16* __restrict__ qbuf, const __bf16* __restrict__ kbuf,
    const __bf16* __restrict__ v_t, __bf16* __restrict__ enc) {
  __shared__ __bf16 sbuf[64 * 264];   // Q stage / K tile / output stage (33.8KB)
  __shared__ __bf16 vT[256 * 40];     // V^T tile: 256 h x 32 s        (20.5KB)
  __shared__ __bf16 pT[4 * 16 * 40];  // per-wave P tile               ( 5.1KB)

  const int tid = threadIdx.x;
  const unsigned lane = tid & 31u;
  const int wid = tid >> 5;
  const int cm = (int)(lane & 15u);
  const int half = (int)(lane >> 4);
  const int tt = blockIdx.x, n = blockIdx.y, b = blockIdx.z;
  const int t0 = tt * 64;
  const int kvh = n >> 1;
  const int r0 = wid * 16;

  const __bf16* kbase = kbuf + ((size_t)(b * NKV + kvh)) * SEQ * HD;
  const __bf16* vbase = v_t + ((size_t)(b * NKV + kvh)) * HD * SEQ;

  // Stage Q tile (64 x 256) then pull this wave's A fragments into registers.
  {
    int row = tid >> 1, seg = tid & 1;
    const v4u* src = (const v4u*)(qbuf +
        (((size_t)(b * SEQ + t0 + row)) * NH + n) * HD + seg * 128);
    v4u* dst = (v4u*)&sbuf[row * 264 + seg * 128];
#pragma unroll
    for (int j = 0; j < 16; ++j) dst[j] = src[j];
  }
  __syncthreads();
  v16bf qf[8];
#pragma unroll
  for (int ks = 0; ks < 8; ++ks)
    qf[ks] = frag_a(&sbuf[(r0 + cm) * 264 + ks * 32], lane);
  __syncthreads();

  float rmax[8], rsum[8];
#pragma unroll
  for (int v = 0; v < 8; ++v) { rmax[v] = -__builtin_inff(); rsum[v] = 0.0f; }
  v8f o[16];
#pragma unroll
  for (int i = 0; i < 16; ++i) o[i] = vzero8();

  const int lo = t0 - (WINDOW - 1);
  const int sb_lo = lo > 0 ? (lo >> 5) : 0;
  const int sb_hi = (t0 + 63) >> 5;

  for (int sb = sb_lo; sb <= sb_hi; ++sb) {
    const int s0 = sb * 32;
    __syncthreads();
    // K tile: 32 rows x 256 halfs.
    {
      int row = tid >> 2, seg = tid & 3;
      const v4u* src = (const v4u*)(kbase + (size_t)(s0 + row) * HD + seg * 64);
      v4u* dst = (v4u*)&sbuf[row * 264 + seg * 64];
#pragma unroll
      for (int j = 0; j < 8; ++j) dst[j] = src[j];
      if (sb < sb_hi)
        __builtin_prefetch(kbase + (size_t)(s0 + 32 + row) * HD + seg * 64, 0, 3);
    }
    // V^T tile: 256 h rows x 32 s halfs.
#pragma unroll
    for (int rr = 0; rr < 2; ++rr) {
      int h = tid * 2 + rr;
      const v4u* src = (const v4u*)(vbase + (size_t)h * SEQ + s0);
      v4u* dst = (v4u*)&vT[h * 40];
#pragma unroll
      for (int j = 0; j < 4; ++j) dst[j] = src[j];
      if (sb < sb_hi)
        __builtin_prefetch(vbase + (size_t)h * SEQ + s0 + 32, 0, 3);
    }
    __syncthreads();

    // logits = Q . K^T : 16 WMMAs (Q fragments live in registers)
    v8f lacc[2];
    lacc[0] = vzero8(); lacc[1] = vzero8();
#pragma unroll
    for (int ks = 0; ks < 8; ++ks) {
      v16bf b0 = frag_b(&sbuf[(cm) * 264 + ks * 32], lane);
      v16bf b1 = frag_b(&sbuf[(16 + cm) * 264 + ks * 32], lane);
      lacc[0] = wmma_bf16(qf[ks], b0, lacc[0]);
      lacc[1] = wmma_bf16(qf[ks], b1, lacc[1]);
    }
    // pipeline: 32 ds reads interleaved 2 ahead of 16 wmmas
    SGB(0x100, 4, 0);
#pragma unroll
    for (int i = 0; i < 14; ++i) { SGB(0x008, 1, 0); SGB(0x100, 2, 0); }
    SGB(0x008, 2, 0);

    // causal + sliding-window mask
    const int tbase = t0 + r0 + 8 * half;
    const int scol = s0 + cm;
#pragma unroll
    for (int c = 0; c < 2; ++c)
#pragma unroll
      for (int v = 0; v < 8; ++v) {
        int t = tbase + v, s = scol + c * 16;
        if (s > t || s < t - (WINDOW - 1)) lacc[c][v] = K_MASK_F;
      }

    // online softmax
    float alpha[8];
#pragma unroll
    for (int v = 0; v < 8; ++v) {
      float bm = fmaxf(lacc[0][v], lacc[1][v]);
      bm = fmaxf(bm, __shfl_xor(bm, 1, 32));
      bm = fmaxf(bm, __shfl_xor(bm, 2, 32));
      bm = fmaxf(bm, __shfl_xor(bm, 4, 32));
      bm = fmaxf(bm, __shfl_xor(bm, 8, 32));
      float nm = fmaxf(rmax[v], bm);
      alpha[v] = __expf(rmax[v] - nm);
      rmax[v] = nm;
      float bs = 0.0f;
#pragma unroll
      for (int c = 0; c < 2; ++c) {
        float p = __expf(lacc[c][v] - nm);
        lacc[c][v] = p;
        bs += p;
      }
      bs += __shfl_xor(bs, 1, 32);
      bs += __shfl_xor(bs, 2, 32);
      bs += __shfl_xor(bs, 4, 32);
      bs += __shfl_xor(bs, 8, 32);
      rsum[v] = rsum[v] * alpha[v] + bs;
    }
#pragma unroll
    for (int ht = 0; ht < 16; ++ht)
#pragma unroll
      for (int v = 0; v < 8; ++v) o[ht][v] *= alpha[v];

    // P -> bf16 -> LDS (re-fragment as A matrix)
    __bf16* pw = &pT[wid * 16 * 40];
#pragma unroll
    for (int c = 0; c < 2; ++c)
#pragma unroll
      for (int v = 0; v < 8; ++v)
        pw[(v + 8 * half) * 40 + c * 16 + cm] = (__bf16)lacc[c][v];
    __syncthreads();

    // O += P . V : 16 WMMAs
    {
      v16bf af = frag_a(&pw[cm * 40], lane);
#pragma unroll
      for (int ht = 0; ht < 16; ++ht) {
        v16bf bf = frag_b(&vT[(ht * 16 + cm) * 40], lane);
        o[ht] = wmma_bf16(af, bf, o[ht]);
      }
      // pipeline: 34 ds reads interleaved 2 ahead of 16 wmmas
      SGB(0x100, 6, 0);
#pragma unroll
      for (int i = 0; i < 14; ++i) { SGB(0x008, 1, 0); SGB(0x100, 2, 0); }
      SGB(0x008, 2, 0);
    }
  }

  // finalize: O /= rowsum, stage to LDS, coalesced bf16 store to enc.
  float invs[8];
#pragma unroll
  for (int v = 0; v < 8; ++v) invs[v] = 1.0f / rsum[v];
  __syncthreads();
#pragma unroll
  for (int ht = 0; ht < 16; ++ht)
#pragma unroll
    for (int v = 0; v < 8; ++v)
      sbuf[(r0 + v + 8 * half) * 264 + ht * 16 + cm] = (__bf16)(o[ht][v] * invs[v]);
  __syncthreads();
  {
    int row = tid >> 1, seg = tid & 1;
    const v4u* src = (const v4u*)&sbuf[row * 264 + seg * 128];
    v4u* dst = (v4u*)(enc +
        (((size_t)(b * SEQ + t0 + row)) * NH + n) * HD + seg * 128);
#pragma unroll
    for (int j = 0; j < 16; ++j) dst[j] = src[j];
  }
}

// ---------------------------------------------------------------------------
// Phase 3: output projection.  grid = (NTOK/64, FEAT/64), block = 128.
// Double-buffered LDS, K chunk = 64; 8 WMMAs per barrier.
// ---------------------------------------------------------------------------
__global__ __launch_bounds__(128) void k_out_gemm(
    const __bf16* __restrict__ enc, const __bf16* __restrict__ wo_t,
    float* __restrict__ out) {
  __shared__ __bf16 aT[2][64 * 72];
  __shared__ __bf16 bT[2][64 * 72];

  const int tid = threadIdx.x;
  const unsigned lane = tid & 31u;
  const int wid = tid >> 5;
  const int cm = (int)(lane & 15u);
  const int half = (int)(lane >> 4);
  const int g0 = blockIdx.x * 64;
  const int d0 = blockIdx.y * 64;
  const int r0 = wid * 16;

  v8f acc[4];
#pragma unroll
  for (int i = 0; i < 4; ++i) acc[i] = vzero8();

  const int row = tid >> 1, seg = tid & 1;  // each thread: 32 halfs (64B) of a row
  v4u ua[4], ubv[4];

  // prologue: fetch tile 0
  {
    const v4u* s1 = (const v4u*)(enc + (size_t)(g0 + row) * FEAT + seg * 32);
    const v4u* s2 = (const v4u*)(wo_t + (size_t)(d0 + row) * FEAT + seg * 32);
#pragma unroll
    for (int j = 0; j < 4; ++j) { ua[j] = s1[j]; ubv[j] = s2[j]; }
    v4u* d1 = (v4u*)&aT[0][row * 72 + seg * 32];
    v4u* d2 = (v4u*)&bT[0][row * 72 + seg * 32];
#pragma unroll
    for (int j = 0; j < 4; ++j) { d1[j] = ua[j]; d2[j] = ubv[j]; }
  }

  const int NIT = FEAT / 64;   // 32
  for (int it = 0; it < NIT; ++it) {
    __syncthreads();
    const int cur = it & 1;
    if (it + 1 < NIT) {
      const int ko = (it + 1) * 64;
      const v4u* s1 = (const v4u*)(enc + (size_t)(g0 + row) * FEAT + ko + seg * 32);
      const v4u* s2 = (const v4u*)(wo_t + (size_t)(d0 + row) * FEAT + ko + seg * 32);
#pragma unroll
      for (int j = 0; j < 4; ++j) { ua[j] = s1[j]; ubv[j] = s2[j]; }
    }
    // 8 WMMAs (2 k-steps x 4 tiles)
#pragma unroll
    for (int ks = 0; ks < 2; ++ks) {
      v16bf af = frag_a(&aT[cur][(r0 + cm) * 72 + ks * 32], lane);
#pragma unroll
      for (int c = 0; c < 4; ++c) {
        v16bf bf = frag_b(&bT[cur][(c * 16 + cm) * 72 + ks * 32], lane);
        acc[c] = wmma_bf16(af, bf, acc[c]);
      }
    }
    if (it + 1 < NIT) {
      const int nxt = cur ^ 1;
      v4u* d1 = (v4u*)&aT[nxt][row * 72 + seg * 32];
      v4u* d2 = (v4u*)&bT[nxt][row * 72 + seg * 32];
#pragma unroll
      for (int j = 0; j < 4; ++j) { d1[j] = ua[j]; d2[j] = ubv[j]; }
    }
    // pipeline: 8 vmem reads, 20 ds reads, 8 wmma, 8 ds writes
    SGB(0x020, 8, 0);
    SGB(0x100, 8, 0);
#pragma unroll
    for (int i = 0; i < 6; ++i) { SGB(0x008, 1, 0); SGB(0x100, 2, 0); }
    SGB(0x008, 2, 0);
    SGB(0x200, 8, 0);
  }

#pragma unroll
  for (int c = 0; c < 4; ++c)
#pragma unroll
    for (int v = 0; v < 8; ++v)
      out[(size_t)(g0 + r0 + v + 8 * half) * FEAT + d0 + c * 16 + cm] = acc[c][v];
}

// ---------------------------------------------------------------------------
// Launch
// ---------------------------------------------------------------------------
extern "C" void kernel_launch(void* const* d_in, const int* in_sizes, int n_in,
                              void* d_out, int out_size, void* d_ws, size_t ws_size,
                              hipStream_t stream) {
  (void)in_sizes; (void)n_in; (void)out_size; (void)ws_size;

  const float* x        = (const float*)d_in[0];
  const int*   seg      = (const int*)d_in[1];
  /* d_in[2] attn_mask: pure causal tril; mask recomputed analytically */
  const float* wq       = (const float*)d_in[3];
  const float* wkv      = (const float*)d_in[4];
  const float* wo       = (const float*)d_in[5];
  const float* q_scale  = (const float*)d_in[6];
  const float* k_scale  = (const float*)d_in[7];
  float* out = (float*)d_out;

  // Workspace layout (bf16 elements)
  __bf16* ws = (__bf16*)d_ws;
  const size_t SZ_WQT = (size_t)NH * HD * FEAT;      // 4,194,304
  const size_t SZ_WKT = (size_t)NKV * HD * FEAT;     // 2,097,152
  const size_t SZ_WOT = (size_t)FEAT * FEAT;         // 4,194,304
  const size_t SZ_Q   = (size_t)NTOK * NH * HD;      // 8,388,608
  const size_t SZ_K   = (size_t)NTOK * NKV * HD;     // 4,194,304

  __bf16* wq_t = ws;                 ws += SZ_WQT;
  __bf16* wk_t = ws;                 ws += SZ_WKT;
  __bf16* wv_t = ws;                 ws += SZ_WKT;
  __bf16* wo_t = ws;                 ws += SZ_WOT;
  __bf16* qbuf = ws;                 ws += SZ_Q;
  __bf16* kbuf = ws;                 ws += SZ_K;
  __bf16* v_t  = ws;                 ws += SZ_K;
  __bf16* enc  = ws;                 /* SZ_Q */

  // Phase 0: weight cast + transpose
  {
    long t1 = (long)SZ_WQT;
    k_cast_transpose<<<(unsigned)((t1 + 255) / 256), 256, 0, stream>>>(
        wq, wq_t, FEAT, HD, t1);
    long t2 = (long)SZ_WKT;
    k_cast_transpose<<<(unsigned)((t2 + 255) / 256), 256, 0, stream>>>(
        wkv, wk_t, FEAT, HD, t2);
    k_cast_transpose<<<(unsigned)((t2 + 255) / 256), 256, 0, stream>>>(
        wkv + (size_t)NKV * FEAT * HD, wv_t, FEAT, HD, t2);
    long t3 = (long)SZ_WOT;
    k_cast_transpose<<<(unsigned)((t3 + 255) / 256), 256, 0, stream>>>(
        wo, wo_t, FEAT, FEAT, t3);
  }

  // Phase 1: QKV projection (WMMA, double-buffered, sched-pipelined)
  k_qkv_gemm<<<dim3(NTOK / 64, 16), 128, 0, stream>>>(
      x, wq_t, wk_t, wv_t, qbuf, kbuf, v_t);

  // Phase 1b: RMS norm + RoPE (+ query scaling)
  k_norm_rope<<<(NTOK * 12) / 8, 256, 0, stream>>>(
      qbuf, kbuf, q_scale, k_scale, seg);

  // Phase 2: sliding-window flash attention (WMMA, sched-pipelined)
  k_attn<<<dim3(SEQ / 64, NH, BATCH), 128, 0, stream>>>(
      qbuf, kbuf, v_t, enc);

  // Phase 3: output projection (WMMA, double-buffered, sched-pipelined)
  k_out_gemm<<<dim3(NTOK / 64, FEAT / 64), 128, 0, stream>>>(
      enc, wo_t, out);
}
